// Temper_5729486372964
// MI455X (gfx1250) — matrix operand order repaired and checked by reading, not compiled, not run on hardware
//
#include <hip/hip_runtime.h>
#include <stdint.h>

#define HIDDEN  2048
#define BATCH   16384
#define NUM_OPS 3

#define BM 128          // M tile (rows)
#define BN 128          // N tile (cols)
#define BK 32           // K tile == WMMA K
#define LDA 40          // padded LDS K-stride (bf16 elems): 64B row + 16B pad
#define LDB 40
#define NTILES     (HIDDEN / BN)            // 16
#define MAX_MTILES (BATCH / BM + NUM_OPS)   // 131 (per-expert round-up padding)
#define PAD_SLOTS  (MAX_MTILES * BM)        // 16768

typedef __bf16 bf16_t;
typedef bf16_t v16bf __attribute__((ext_vector_type(16)));
typedef bf16_t v2bf  __attribute__((ext_vector_type(2)));
typedef float  v8f   __attribute__((ext_vector_type(8)));
typedef float  v2f   __attribute__((ext_vector_type(2)));
typedef int    v4i   __attribute__((ext_vector_type(4)));
typedef int    v8i   __attribute__((ext_vector_type(8)));

union Frag { v16bf v; uint4 q[2]; };

// f32 -> bf16 native converts (v_cvt_pk_bf16_f32 on gfx1250)
__device__ __forceinline__ unsigned short f2bf(float f) {
    union { bf16_t b; unsigned short u; } cv;
    cv.b = (bf16_t)f;
    return cv.u;
}
__device__ __forceinline__ unsigned pack2bf(float a, float b) {
    v2f f = {a, b};
    union { v2bf v; unsigned u; } cv;
    cv.v = __builtin_convertvector(f, v2bf);
    return cv.u;
}

// ---------------- bucketing kernels ----------------
// meta layout (ints): [0..2] counts, [4..6] cursors, [8..11] padded offsets (offP[3]=total)
__global__ void k_init(int* meta) { if (threadIdx.x < 12) meta[threadIdx.x] = 0; }

__global__ void k_count(const int* __restrict__ ops, int* __restrict__ meta) {
    int b = blockIdx.x * blockDim.x + threadIdx.x;
    if (b < BATCH) atomicAdd(&meta[ops[b]], 1);
}

__global__ void k_offsets(int* meta) {
    int c0 = meta[0], c1 = meta[1], c2 = meta[2];
    meta[8]  = 0;
    meta[9]  = ((c0 + BM - 1) / BM) * BM;
    meta[10] = meta[9]  + ((c1 + BM - 1) / BM) * BM;
    meta[11] = meta[10] + ((c2 + BM - 1) / BM) * BM;
}

__global__ void k_fill(int* __restrict__ rowOf) {
    int i = blockIdx.x * blockDim.x + threadIdx.x;
    if (i < PAD_SLOTS) rowOf[i] = -1;
}

__global__ void k_assign(const int* __restrict__ ops, int* __restrict__ meta,
                         int* __restrict__ rowOf) {
    int b = blockIdx.x * blockDim.x + threadIdx.x;
    if (b < BATCH) {
        int e = ops[b];
        int p = atomicAdd(&meta[4 + e], 1);
        rowOf[meta[8 + e] + p] = b;
    }
}

// ---------------- fused bucketed GEMM + bias + ReLU ----------------
// LAYER 1: A = gathered x rows (f32 -> bf16), writes bf16 h to hbuf (slot space)
// LAYER 2: A tiles double-buffered via the Tensor Data Mover; scatters f32 out
template <int LAYER>
__global__ __launch_bounds__(256)
void moe_gemm(const float* __restrict__ x, const float* __restrict__ W,
              const float* __restrict__ bias, const int* __restrict__ offP,
              const int* __restrict__ rowOf, unsigned short* __restrict__ hbuf,
              float* __restrict__ out)
{
    __shared__ unsigned short sA[(LAYER == 2 ? 2 : 1) * BM * LDA]; // [buf][m][k] bf16
    __shared__ unsigned short sB[BN * LDB];                        // [n][k] bf16 (transposed)
    __shared__ int sRow[BM];

    const int tid      = threadIdx.x;
    const int slotBase = blockIdx.y * BM;
    if (slotBase >= offP[3]) return;          // beyond padded slot space

    int e = 0;                                 // expert owning this M tile
    if (slotBase >= offP[2]) e = 2; else if (slotBase >= offP[1]) e = 1;

    const int n0 = blockIdx.x * BN;

    if (tid < BM) sRow[tid] = rowOf[slotBase + tid];
    __syncthreads();

    const int lane   = tid & 31;
    const int wave   = tid >> 5;
    const int waveM  = wave & 3;              // 4 waves along M
    const int waveN  = wave >> 2;             // 2 waves along N
    const int mrow   = lane & 15;
    const int hiHalf = lane >> 4;             // 0/1 half-wave

    v8f acc[2][4];
    v8f zero = {};
    #pragma unroll
    for (int sm = 0; sm < 2; ++sm)
        #pragma unroll
        for (int sn = 0; sn < 4; ++sn) acc[sm][sn] = zero;

    const float* Wbase = W + (size_t)e * HIDDEN * HIDDEN;  // W[e][k][n], k = contraction

    // ---- K-invariant per-thread addresses ----
    // Layer-1 A loader: 32 rows x 8 float4-chunks per pass, 4 passes.
    // Masked (padding) rows just read row 0: their outputs are masked at store.
    const int aRow = tid >> 3;            // 0..31
    const int aKq  = (tid & 7) * 4;       // f32 quad within K tile
    const float* xrow[4];
    if constexpr (LAYER == 1) {
        #pragma unroll
        for (int p = 0; p < 4; ++p) {
            int gr = sRow[aRow + p * 32];
            xrow[p] = x + (size_t)(gr < 0 ? 0 : gr) * HIDDEN + aKq;
        }
    }
    // B loader: 32 n-quads x (2 consecutive k-rows) per pass, 2 passes.
    const int bCol = (tid & 31) * 4;      // n quad
    const int bK2  = (tid >> 5) * 2;      // k pair: 0,2,...,14
    const float* wcol = Wbase + (size_t)bK2 * HIDDEN + n0 + bCol;

    // Tensor Data Mover descriptor (2D, bf16, padded LDS rows), layer 2 only.
    v8i g1; unsigned ldsA = 0; uint64_t gaBase = 0;
    if constexpr (LAYER == 2) {
        ldsA   = (unsigned)(uintptr_t)&sA[0];
        gaBase = (uint64_t)(uintptr_t)(hbuf + (size_t)slotBase * HIDDEN);
        g1[0] = (1 << 16)      // data_size = 2 bytes
              | (1 << 20)      // pad_enable
              | (3 << 22)      // pad_interval: 16 DWORDs (64B) per row
              | (3 << 25);     // pad_amount:   4 DWORDs (16B) -> LDA = 40 elems
        g1[1] = (int)(HIDDEN << 16);               // tensor_dim0 lo16 @ bits63:48
        g1[2] = (int)((PAD_SLOTS & 0xFFFF) << 16); // dim0 hi=0 | tensor_dim1 lo
        g1[3] = (BK << 16);                        // tensor_dim1 hi=0 | tile_dim0=32
        g1[4] = BM;                                // tile_dim1=128, tile_dim2=0
        g1[5] = HIDDEN;                            // tensor_dim0_stride (elems)
        g1[6] = 0;
        g1[7] = 0;
    }
    auto tdm_issue = [&](int kNext, int bufSel) {
        uint64_t ga = gaBase + (uint64_t)kNext * 2;
        v4i g0;
        g0[0] = 1;                                             // count = 1 valid D#
        g0[1] = (int)(ldsA + (unsigned)bufSel * (BM * LDA * 2)); // lds_addr
        g0[2] = (int)(unsigned)(ga & 0xFFFFFFFFu);             // global_addr[31:0]
        g0[3] = (int)(((unsigned)(ga >> 32)) & 0x01FFFFFFu)
              | (int)(2u << 30);                               // addr[56:32] | type=2
        asm volatile("tensor_load_to_lds %0, %1" :: "s"(g0), "s"(g1) : "memory");
    };

    if constexpr (LAYER == 2) {
        if (wave == 0) tdm_issue(0, 0);          // prime the pipeline
    }

    int buf = 0;
    #pragma unroll 2
    for (int k0 = 0; k0 < HIDDEN; k0 += BK) {
        // ---- stage A tile (layer 1) / issue next TDM tile (layer 2) ----
        if constexpr (LAYER == 1) {
            #pragma unroll
            for (int p = 0; p < 4; ++p) {
                float4 v = *(const float4*)(xrow[p] + k0);
                __builtin_prefetch(xrow[p] + k0 + BK, 0, 1);   // speculative-safe
                uint2 pk;
                pk.x = pack2bf(v.x, v.y);
                pk.y = pack2bf(v.z, v.w);
                *(uint2*)&sA[(aRow + p * 32) * LDA + aKq] = pk;
            }
        } else {
            if (k0 + BK < HIDDEN) {
                if (wave == 0) tdm_issue(k0 + BK, buf ^ 1);    // overlap next DMA
            }
        }
        // ---- stage B tile: W[k0..+31][n0..+127] -> sB[n][k], packed k-pairs ----
        {
            const float* wp0 = wcol + (size_t)k0 * HIDDEN;
            #pragma unroll
            for (int p = 0; p < 2; ++p) {
                int kk = bK2 + p * 16;
                const float* wp = wp0 + (size_t)(p * 16) * HIDDEN;
                float4 v0 = *(const float4*)wp;
                float4 v1 = *(const float4*)(wp + HIDDEN);
                __builtin_prefetch(wp + (size_t)BK * HIDDEN, 0, 1);
                *(unsigned*)&sB[(bCol + 0) * LDB + kk] = pack2bf(v0.x, v1.x);
                *(unsigned*)&sB[(bCol + 1) * LDB + kk] = pack2bf(v0.y, v1.y);
                *(unsigned*)&sB[(bCol + 2) * LDB + kk] = pack2bf(v0.z, v1.z);
                *(unsigned*)&sB[(bCol + 3) * LDB + kk] = pack2bf(v0.w, v1.w);
            }
        }
        if constexpr (LAYER == 2) {
            if (k0 + BK < HIDDEN) {
                if (wave == 0) __builtin_amdgcn_s_wait_tensorcnt(1); // tile k0 landed
            } else {
                if (wave == 0) __builtin_amdgcn_s_wait_tensorcnt(0);
            }
        }
        __syncthreads();

        // ---- build fragments per documented 16-bit WMMA layouts ----
        const unsigned short* sAb = sA + (LAYER == 2 ? buf * (BM * LDA) : 0);
        Frag a[2], b[4];
        #pragma unroll
        for (int sm = 0; sm < 2; ++sm) {
            int m  = waveM * 32 + sm * 16 + mrow;
            int kb = hiHalf * 8;                   // lanes 0-15: K 0..7/16..23; hi: +8
            a[sm].q[0] = *(const uint4*)&sAb[m * LDA + kb];
            a[sm].q[1] = *(const uint4*)&sAb[m * LDA + 16 + kb];
        }
        #pragma unroll
        for (int sn = 0; sn < 4; ++sn) {
            int n  = waveN * 64 + sn * 16 + mrow;  // column = lane%16
            int kb = hiHalf * 8;                   // lanes 0-15: K 0..7/16..23; hi: +8
            b[sn].q[0] = *(const uint4*)&sB[n * LDB + kb];
            b[sn].q[1] = *(const uint4*)&sB[n * LDB + 16 + kb];
        }
        #pragma unroll
        for (int sm = 0; sm < 2; ++sm)
            #pragma unroll
            for (int sn = 0; sn < 4; ++sn)
                acc[sm][sn] = __builtin_amdgcn_wmma_f32_16x16x32_bf16(
                    false, a[sm].v, false, b[sn].v,
                    (short)0, acc[sm][sn], false, false);
        __syncthreads();
        buf ^= 1;
    }

    // ---- epilogue: C layout => M = r + 8*hiHalf, N = lane%16 ----
    #pragma unroll
    for (int sm = 0; sm < 2; ++sm) {
        #pragma unroll
        for (int sn = 0; sn < 4; ++sn) {
            int nloc = waveN * 64 + sn * 16 + mrow;
            float bv = bias[(size_t)e * HIDDEN + n0 + nloc];
            #pragma unroll
            for (int r = 0; r < 8; ++r) {
                int mloc = waveM * 32 + sm * 16 + r + hiHalf * 8;
                int gr   = sRow[mloc];
                if (gr < 0) continue;
                float val = acc[sm][sn][r] + bv;
                val = fmaxf(val, 0.0f);
                if constexpr (LAYER == 1)
                    hbuf[(size_t)(slotBase + mloc) * HIDDEN + n0 + nloc] = f2bf(val);
                else
                    out[(size_t)gr * HIDDEN + n0 + nloc] = val;
            }
        }
    }
}

// ---------------- host entry ----------------
extern "C" void kernel_launch(void* const* d_in, const int* in_sizes, int n_in,
                              void* d_out, int out_size, void* d_ws, size_t ws_size,
                              hipStream_t stream) {
    const float* x  = (const float*)d_in[0];
    const float* W1 = (const float*)d_in[1];
    const float* b1 = (const float*)d_in[2];
    const float* W2 = (const float*)d_in[3];
    const float* b2 = (const float*)d_in[4];
    const int*  ops = (const int*)d_in[5];
    float* out = (float*)d_out;

    int* meta  = (int*)d_ws;                                    // 16 ints
    int* rowOf = meta + 16;                                     // PAD_SLOTS ints
    unsigned short* hbuf = (unsigned short*)((char*)d_ws + (1 << 17)); // bf16 h

    k_init   <<<1, 32, 0, stream>>>(meta);
    k_count  <<<BATCH / 256, 256, 0, stream>>>(ops, meta);
    k_offsets<<<1, 1, 0, stream>>>(meta);
    k_fill   <<<(PAD_SLOTS + 255) / 256, 256, 0, stream>>>(rowOf);
    k_assign <<<BATCH / 256, 256, 0, stream>>>(ops, meta, rowOf);

    dim3 grid(NTILES, MAX_MTILES);
    moe_gemm<1><<<grid, 256, 0, stream>>>(x,       W1, b1, meta + 8, rowOf, hbuf, nullptr);
    moe_gemm<2><<<grid, 256, 0, stream>>>(nullptr, W2, b2, meta + 8, rowOf, hbuf, out);
}
